// Attention_4380866642117
// MI455X (gfx1250) — matrix operand order repaired and checked by reading, not compiled, not run on hardware
//
#include <hip/hip_runtime.h>
#include <hip/hip_bf16.h>
#include <math.h>

// Problem constants (reference: b=8, lq=lk=2048, d=256)
#define B_   8
#define LQ_  2048
#define LK_  2048
#define D_   256
#define LN_EPS_   1e-5f
#define MASK_FILL_ (-10000000000.0f)

typedef __attribute__((ext_vector_type(16))) __bf16          v16bf;
typedef __attribute__((ext_vector_type(8)))  float           v8f;
typedef __attribute__((ext_vector_type(8)))  unsigned short  us8;
typedef __attribute__((ext_vector_type(4)))  float           f4;

union BF16Frag {
    v16bf v;
    unsigned short u[16];
    us8 h[2];
};

__device__ __forceinline__ unsigned short f2bf(float f) {
    unsigned int u = __float_as_uint(f);
    u += 0x7FFFu + ((u >> 16) & 1u);   // round-to-nearest-even
    return (unsigned short)(u >> 16);
}

// ---------------------------------------------------------------------------
// 1) sq[b,i] = q[b,i,:].w_q ; sk[b,j] = k[b,j,:].w_k   (one wave32 per row)
// ---------------------------------------------------------------------------
__global__ void k_scores(const float* __restrict__ q, const float* __restrict__ k,
                         const float* __restrict__ sa,
                         float* __restrict__ sq, float* __restrict__ sk) {
    int wave = (int)((blockIdx.x * blockDim.x + threadIdx.x) >> 5);
    int lane = threadIdx.x & 31;
    const int total_q = B_ * LQ_;
    const float* row; const float* w; float* out; int idx;
    if (wave < total_q) { row = q + (size_t)wave * D_;            w = sa;      out = sq; idx = wave; }
    else { int r = wave - total_q; row = k + (size_t)r * D_;      w = sa + D_; out = sk; idx = r; }
    float s = 0.f;
    int c = lane * 8;
#pragma unroll
    for (int j = 0; j < 8; ++j) s += row[c + j] * w[c + j];
#pragma unroll
    for (int off = 16; off > 0; off >>= 1) s += __shfl_xor(s, off, 32);
    if (lane == 0) out[idx] = s;
}

// ---------------------------------------------------------------------------
// 2) vT[b, d, lk] = bf16(v[b, lk, d])  — 32x32 LDS tile transpose
// ---------------------------------------------------------------------------
__global__ void k_transpose_v(const float* __restrict__ v, unsigned short* __restrict__ vt) {
    __shared__ float tile[32][33];
    int b  = blockIdx.z;
    int r0 = blockIdx.x * 32;            // lk base
    int c0 = blockIdx.y * 32;            // d base
    int tx = threadIdx.x & 31, ty = threadIdx.x >> 5;  // 32 x 8
    const float* src = v + (size_t)b * LK_ * D_;
#pragma unroll
    for (int i = 0; i < 4; ++i) {
        int r = r0 + ty + i * 8;
        tile[ty + i * 8][tx] = src[(size_t)r * D_ + c0 + tx];
    }
    __syncthreads();
    unsigned short* dst = vt + (size_t)b * D_ * LK_;
#pragma unroll
    for (int i = 0; i < 4; ++i) {
        int c = c0 + ty + i * 8;
        dst[(size_t)c * LK_ + r0 + tx] = f2bf(tile[tx][ty + i * 8]);
    }
}

// ---------------------------------------------------------------------------
// 3) fc_w -> bf16 (layout kept (d_out, d_in) row-major: B-frag rows contiguous)
// ---------------------------------------------------------------------------
__global__ void k_cvt_fc(const float* __restrict__ w, unsigned short* __restrict__ wb) {
    int i = blockIdx.x * blockDim.x + threadIdx.x;
    wb[i] = f2bf(w[i]);
}

// ---------------------------------------------------------------------------
// 4) Fused masked-softmax + (P @ V) WMMA GEMM.
//    Block = 256 threads (8 waves) owns a 128-row tile of one batch:
//      phase 1: chunked online softmax -> per-row (max, 1/sum), mask pass #1
//      phase 2: regenerate p = exp(v-m)*inv per 16x32 A-fragment, store f32
//               attn row to d_out AND run v_wmma_f32_16x16x32_bf16 against
//               bf16 V^T tiles; accumulate the full 256-wide output.
//    sk row is staged to LDS with CDNA5 async loads (ASYNCcnt).
// ---------------------------------------------------------------------------
__global__ void __launch_bounds__(256, 1)
k_attn_pv(const float* __restrict__ sq, const float* __restrict__ sk,
          const unsigned char* __restrict__ mask,
          const unsigned short* __restrict__ vt,
          float* __restrict__ attn, unsigned short* __restrict__ out1) {
    __shared__ float s_sk[LK_];      // 8 KB: sk[b,:]
    __shared__ float s_m[256];
    __shared__ float s_s[256];
    __shared__ float s_rowm[128];
    __shared__ float s_rowinv[128];

    int b    = blockIdx.y;
    int row0 = blockIdx.x * 128;     // block-local row window
    int tid  = threadIdx.x;

    // ---- stage sk[b,:] into LDS via async global->LDS DMA (2 x 16B / thread)
    {
        const float* skb = sk + b * LK_;
        unsigned lds0 = (unsigned)(uintptr_t)(&s_sk[0]);
#pragma unroll
        for (int it = 0; it < 2; ++it) {
            int foff = (tid + it * 256) * 4;                    // float index
            unsigned lds = lds0 + (unsigned)foff * 4u;          // byte addr
            unsigned long long ga = (unsigned long long)(uintptr_t)(skb + foff);
            asm volatile("global_load_async_to_lds_b128 %0, %1, off"
                         :: "v"(lds), "v"(ga) : "memory");
        }
        asm volatile("s_wait_asynccnt 0" ::: "memory");
    }
    __syncthreads();

    // ---- phase 1: online softmax stats; 2 threads per row, 1024 cols each
    {
        int rt    = tid >> 1;
        int jbase = (tid & 1) * 1024;
        int grow  = b * LQ_ + row0 + rt;
        float qi  = sq[grow];
        const unsigned char* mrow = mask + (size_t)grow * LK_;
        float m = -3.4e38f, ssum = 0.f;
        for (int jc = jbase; jc < jbase + 1024; jc += 16) {
            __builtin_prefetch(mrow + jc + 256, 0, 1);
            uint4 mm = *(const uint4*)(mrow + jc);
            unsigned mw[4] = {mm.x, mm.y, mm.z, mm.w};
            f4 k0v = *(const f4*)(s_sk + jc);
            f4 k1v = *(const f4*)(s_sk + jc + 4);
            f4 k2v = *(const f4*)(s_sk + jc + 8);
            f4 k3v = *(const f4*)(s_sk + jc + 12);
            float vv[16];
            float cmax = -3.4e38f;
#pragma unroll
            for (int e = 0; e < 4; ++e) {
                vv[e]      = ((mw[0] >> (8 * e)) & 0xffu) ? MASK_FILL_ : (qi + k0v[e]);
                vv[4 + e]  = ((mw[1] >> (8 * e)) & 0xffu) ? MASK_FILL_ : (qi + k1v[e]);
                vv[8 + e]  = ((mw[2] >> (8 * e)) & 0xffu) ? MASK_FILL_ : (qi + k2v[e]);
                vv[12 + e] = ((mw[3] >> (8 * e)) & 0xffu) ? MASK_FILL_ : (qi + k3v[e]);
            }
#pragma unroll
            for (int e = 0; e < 16; ++e) cmax = fmaxf(cmax, vv[e]);
            float mn = fmaxf(m, cmax);
            float cs = 0.f;
#pragma unroll
            for (int e = 0; e < 16; ++e) cs += __expf(vv[e] - mn);
            ssum = ssum * __expf(m - mn) + cs;
            m = mn;
        }
        s_m[tid] = m;
        s_s[tid] = ssum;
    }
    __syncthreads();
    if (tid < 128) {
        float m0 = s_m[2 * tid], m1 = s_m[2 * tid + 1];
        float mm = fmaxf(m0, m1);
        float st = s_s[2 * tid] * __expf(m0 - mm) + s_s[2 * tid + 1] * __expf(m1 - mm);
        s_rowm[tid]   = mm;
        s_rowinv[tid] = 1.f / st;
    }
    __syncthreads();

    // ---- phase 2: per-wave 16 rows x 256 cols, K-loop over 2048 in steps of 32
    int wv   = tid >> 5;
    int lane = tid & 31;
    int half = lane >> 4;
    int l16  = lane & 15;
    int lrow = wv * 16 + l16;            // block-local row this lane generates
    int growA = b * LQ_ + row0 + lrow;
    float qiA  = sq[growA];
    float rm   = s_rowm[lrow];
    float rinv = s_rowinv[lrow];
    const unsigned char* mA = mask + (size_t)growA * LK_;
    float* arow = attn + (size_t)growA * LK_;
    int abase = half * 8;    // 16-bit A layout: lane-half K chunks +0 / +8
    int khalf = half * 8;    // 16-bit B layout: lane-half K chunks +0 / +8

    v8f acc[16];
#pragma unroll
    for (int nt = 0; nt < 16; ++nt)
#pragma unroll
        for (int e = 0; e < 8; ++e) acc[nt][e] = 0.f;

    for (int k0 = 0; k0 < LK_; k0 += 32) {
        __builtin_prefetch(mA + k0 + 512, 0, 1);
        int c0 = k0 + abase;
        int c1 = k0 + 16 + abase;
        unsigned long long mb0 = *(const unsigned long long*)(mA + c0);
        unsigned long long mb1 = *(const unsigned long long*)(mA + c1);
        f4 s0 = *(const f4*)(s_sk + c0);
        f4 s1 = *(const f4*)(s_sk + c0 + 4);
        f4 s2 = *(const f4*)(s_sk + c1);
        f4 s3 = *(const f4*)(s_sk + c1 + 4);
        f4 p0, p1, p2, p3;
        BF16Frag a;
#pragma unroll
        for (int e = 0; e < 4; ++e) {
            float v0 = ((mb0 >> (8 * e))        & 0xffull) ? MASK_FILL_ : (qiA + s0[e]);
            float v1 = ((mb0 >> (8 * (e + 4)))  & 0xffull) ? MASK_FILL_ : (qiA + s1[e]);
            float v2 = ((mb1 >> (8 * e))        & 0xffull) ? MASK_FILL_ : (qiA + s2[e]);
            float v3 = ((mb1 >> (8 * (e + 4)))  & 0xffull) ? MASK_FILL_ : (qiA + s3[e]);
            p0[e] = __expf(v0 - rm) * rinv;
            p1[e] = __expf(v1 - rm) * rinv;
            p2[e] = __expf(v2 - rm) * rinv;
            p3[e] = __expf(v3 - rm) * rinv;
            a.u[e]      = f2bf(p0[e]);
            a.u[4 + e]  = f2bf(p1[e]);
            a.u[8 + e]  = f2bf(p2[e]);
            a.u[12 + e] = f2bf(p3[e]);
        }
        *(f4*)(arow + c0)     = p0;     // f32 attention output (d_out region 2)
        *(f4*)(arow + c0 + 4) = p1;
        *(f4*)(arow + c1)     = p2;
        *(f4*)(arow + c1 + 4) = p3;
#pragma unroll
        for (int nt = 0; nt < 16; ++nt) {
            const us8* bp = (const us8*)(vt + (size_t)(b * D_ + nt * 16 + l16) * LK_ + k0 + khalf);
            BF16Frag bf;
            bf.h[0] = bp[0];
            bf.h[1] = bp[2];
            acc[nt] = __builtin_amdgcn_wmma_f32_16x16x32_bf16(
                false, a.v, false, bf.v, (short)0, acc[nt], false, false);
        }
    }

    // ---- epilogue: out1 as bf16 (FC input)
#pragma unroll
    for (int nt = 0; nt < 16; ++nt) {
        int col = nt * 16 + l16;
#pragma unroll
        for (int r = 0; r < 8; ++r) {
            int row = row0 + wv * 16 + r + half * 8;
            out1[(size_t)(b * LQ_ + row) * D_ + col] = f2bf(acc[nt][r]);
        }
    }
}

// ---------------------------------------------------------------------------
// 5) Fused  out = LayerNorm(out1 @ fc_w.T + q) * gamma + beta.
//    Wave owns 16 full rows (16 accumulator tiles = all 256 cols), so the
//    LayerNorm row reduction is a 16-lane shfl_xor within each lane-half.
// ---------------------------------------------------------------------------
__global__ void __launch_bounds__(256, 1)
k_fc_ln(const unsigned short* __restrict__ x1, const unsigned short* __restrict__ wb,
        const float* __restrict__ q, const float* __restrict__ gamma,
        const float* __restrict__ beta, float* __restrict__ out) {
    int wv   = threadIdx.x >> 5;
    int lane = threadIdx.x & 31;
    int half = lane >> 4;
    int l16  = lane & 15;
    int mbase = blockIdx.x * 128 + wv * 16;
    const unsigned short* aptr = x1 + (size_t)(mbase + l16) * D_;
    int abase = half * 8;
    int khalf = half * 8;

    v8f acc[16];
#pragma unroll
    for (int nt = 0; nt < 16; ++nt)
#pragma unroll
        for (int e = 0; e < 8; ++e) acc[nt][e] = 0.f;

#pragma unroll
    for (int k0 = 0; k0 < D_; k0 += 32) {
        BF16Frag a;
        a.h[0] = *(const us8*)(aptr + k0 + abase);
        a.h[1] = *(const us8*)(aptr + k0 + 16 + abase);
#pragma unroll
        for (int nt = 0; nt < 16; ++nt) {
            const us8* bp = (const us8*)(wb + (size_t)(nt * 16 + l16) * D_ + k0 + khalf);
            BF16Frag bf;
            bf.h[0] = bp[0];
            bf.h[1] = bp[1 + 1];
            acc[nt] = __builtin_amdgcn_wmma_f32_16x16x32_bf16(
                false, a.v, false, bf.v, (short)0, acc[nt], false, false);
        }
    }

    // residual add (coalesced per nt across the 16 lanes of each half)
#pragma unroll
    for (int nt = 0; nt < 16; ++nt) {
#pragma unroll
        for (int r = 0; r < 8; ++r) {
            int row = mbase + r + half * 8;
            acc[nt][r] += q[(size_t)row * D_ + nt * 16 + l16];
        }
    }

    // per-row LayerNorm: row lives in one lane-half (fixed r), 16 lanes x 16 nt
#pragma unroll
    for (int r = 0; r < 8; ++r) {
        float s = 0.f, s2 = 0.f;
#pragma unroll
        for (int nt = 0; nt < 16; ++nt) {
            float t = acc[nt][r];
            s += t;
            s2 += t * t;
        }
#pragma unroll
        for (int off = 1; off <= 8; off <<= 1) {
            s  += __shfl_xor(s,  off, 32);
            s2 += __shfl_xor(s2, off, 32);
        }
        float mu  = s  * (1.0f / D_);
        float var = s2 * (1.0f / D_) - mu * mu;
        float inv = rsqrtf(var + LN_EPS_);
        int row = mbase + r + half * 8;
#pragma unroll
        for (int nt = 0; nt < 16; ++nt) {
            int col = nt * 16 + l16;
            out[(size_t)row * D_ + col] = (acc[nt][r] - mu) * inv * gamma[col] + beta[col];
        }
    }
}

// ---------------------------------------------------------------------------
// Host launcher
// ---------------------------------------------------------------------------
extern "C" void kernel_launch(void* const* d_in, const int* in_sizes, int n_in,
                              void* d_out, int out_size, void* d_ws, size_t ws_size,
                              hipStream_t stream) {
    (void)in_sizes; (void)n_in; (void)out_size; (void)ws_size;
    const float* q  = (const float*)d_in[0];
    const float* k  = (const float*)d_in[1];
    const float* v  = (const float*)d_in[2];
    const float* sa = (const float*)d_in[3];
    const float* fw = (const float*)d_in[4];
    const float* g  = (const float*)d_in[5];
    const float* be = (const float*)d_in[6];
    const unsigned char* mask = (const unsigned char*)d_in[7];

    // Workspace carve-up (~16.3 MB total)
    float*          sq   = (float*)d_ws;                             // B*LQ f32
    float*          sk   = sq + (size_t)B_ * LQ_;                    // B*LK f32
    unsigned short* vt   = (unsigned short*)(sk + (size_t)B_ * LK_); // B*D*LK bf16
    unsigned short* fcb  = vt + (size_t)B_ * D_ * LK_;               // D*D bf16
    unsigned short* out1 = fcb + (size_t)D_ * D_;                    // B*LQ*D bf16

    // Output layout: [out (B*LQ*D)] then [attn (B*LQ*LK)], both f32
    float* out_final = (float*)d_out;
    float* attn      = out_final + (size_t)B_ * LQ_ * D_;

    k_scores     <<<(2 * B_ * LQ_) / 8, 256, 0, stream>>>(q, k, sa, sq, sk);
    k_transpose_v<<<dim3(LK_ / 32, D_ / 32, B_), 256, 0, stream>>>(v, vt);
    k_cvt_fc     <<<(D_ * D_) / 256, 256, 0, stream>>>(fw, fcb);
    k_attn_pv    <<<dim3(LQ_ / 128, B_), 256, 0, stream>>>(sq, sk, mask, vt, attn, out1);
    k_fc_ln      <<<(B_ * LQ_) / 128, 256, 0, stream>>>(out1, fcb, q, g, be, out_final);
}
